// AttentionSpace_87729001988510
// MI455X (gfx1250) — matrix operand order repaired and checked by reading, not compiled, not run on hardware
//
#include <hip/hip_runtime.h>

typedef unsigned short u16;
typedef __attribute__((ext_vector_type(16))) __bf16 v16bf;
typedef __attribute__((ext_vector_type(8)))  float  v8f;
typedef __attribute__((ext_vector_type(4)))  int    v4i;

struct __align__(16) U4 { unsigned int x, y, z, w; };

union FragBF16 {
    U4    u4[2];
    v16bf v;
};

__device__ __forceinline__ u16 f2bf(float f) {
    union { float f; unsigned u; } x;
    x.f = f;
    unsigned r = x.u + 0x7FFFu + ((x.u >> 16) & 1u);   // round-to-nearest-even
    return (u16)(r >> 16);
}

// ---- CDNA5 async global->LDS copy (ASYNCcnt path), with sync fallback -----
#if defined(__has_builtin)
#if __has_builtin(__builtin_amdgcn_global_load_async_to_lds_b128)
#define USE_ASYNC_LDS 1
#else
#define USE_ASYNC_LDS 0
#endif
#else
#define USE_ASYNC_LDS 0
#endif

#if USE_ASYNC_LDS
typedef __attribute__((address_space(1))) v4i gas1_v4i;   // global int4
typedef __attribute__((address_space(3))) v4i las3_v4i;   // LDS int4
#endif

__device__ __forceinline__ void copy16_g2l(const void* g, void* l) {
#if USE_ASYNC_LDS
    __builtin_amdgcn_global_load_async_to_lds_b128(
        (gas1_v4i*)g, (las3_v4i*)l, 0, 0);
#else
    *(U4*)l = *(const U4*)g;
#endif
}

__device__ __forceinline__ void wait_async_copies() {
#if USE_ASYNC_LDS
#if __has_builtin(__builtin_amdgcn_s_wait_asynccnt)
    __builtin_amdgcn_s_wait_asynccnt(0);
#else
    asm volatile("s_wait_asynccnt 0x0" ::: "memory");
#endif
#endif
}

// ---------------------------------------------------------------------------
// Kernel 1: fused depthwise conv (K=7, pad=3) producing q,k,v in bf16.
// One block per (b,c) row of length L=4096. x row staged in LDS with halo.
// ---------------------------------------------------------------------------
__global__ void dwconv_qkv_kernel(const float* __restrict__ x,
                                  const float* __restrict__ qw,
                                  const float* __restrict__ kw,
                                  const float* __restrict__ vw,
                                  u16* __restrict__ q,
                                  u16* __restrict__ k,
                                  u16* __restrict__ v) {
    const int L = 4096;
    const int bc = blockIdx.x;          // b*C + c
    const int c  = bc & 1023;

    __shared__ float xs[4096 + 6];

    const float* xrow = x + (size_t)bc * L;
    for (int i = threadIdx.x; i < L; i += 256) xs[i + 3] = xrow[i];
    if (threadIdx.x < 3) { xs[threadIdx.x] = 0.0f; xs[L + 3 + threadIdx.x] = 0.0f; }
    __syncthreads();

    float wq[7], wk[7], wv[7];
#pragma unroll
    for (int t = 0; t < 7; ++t) {
        wq[t] = qw[c * 7 + t];
        wk[t] = kw[c * 7 + t];
        wv[t] = vw[c * 7 + t];
    }

    u16* qr = q + (size_t)bc * L;
    u16* kr = k + (size_t)bc * L;
    u16* vr = v + (size_t)bc * L;

    for (int l = threadIdx.x; l < L; l += 256) {
        float aq = 0.0f, ak = 0.0f, av = 0.0f;
#pragma unroll
        for (int t = 0; t < 7; ++t) {
            float xv = xs[l + t];
            aq = fmaf(xv, wq[t], aq);
            ak = fmaf(xv, wk[t], ak);
            av = fmaf(xv, wv[t], av);
        }
        qr[l] = f2bf(aq);
        kr[l] = f2bf(ak);
        vr[l] = f2bf(av);
    }
}

// ---------------------------------------------------------------------------
// Kernel 2: bf16 tile transpose, src [R x Ccols] -> dst [Ccols x R] per batch.
// 64x64 tiles through padded LDS; both global sides fully coalesced (16B).
// grid = (Ccols/64, R/64, batch)
// ---------------------------------------------------------------------------
__global__ void transpose_bf16_kernel(const u16* __restrict__ src,
                                      u16* __restrict__ dst,
                                      int R, int Ccols) {
    __shared__ __align__(16) u16 tile[64][72];   // +8 pad

    const size_t off = (size_t)blockIdx.z * (size_t)R * (size_t)Ccols;
    const int r0 = blockIdx.y * 64;
    const int c0 = blockIdx.x * 64;

#pragma unroll
    for (int i = 0; i < 2; ++i) {
        int chunk = threadIdx.x + i * 256;       // 512 chunks of 8 elems
        int r = chunk >> 3;
        int cc = (chunk & 7) << 3;
        *(U4*)&tile[r][cc] = *(const U4*)&src[off + (size_t)(r0 + r) * Ccols + c0 + cc];
    }
    __syncthreads();
#pragma unroll
    for (int i = 0; i < 2; ++i) {
        int chunk = threadIdx.x + i * 256;
        int r = chunk >> 3;                      // row within dst tile (= col of src)
        int cc = (chunk & 7) << 3;               // col within dst tile (= row of src)
        __align__(16) u16 tmp[8];
#pragma unroll
        for (int j = 0; j < 8; ++j) tmp[j] = tile[cc + j][r];
        *(U4*)&dst[off + (size_t)(c0 + r) * R + r0 + cc] = *(U4*)tmp;
    }
}

// ---------------------------------------------------------------------------
// Kernel 3: bf16 GEMM, D = scale * (A @ BT^T), f32 accumulate via
// v_wmma_f32_16x16x32_bf16. Both operands K-contiguous:
//   A : [M x K] row-major bf16
//   BT: [N x K] row-major bf16
// Block tile 128x128, BK=32, 256 threads = 8 waves, wave tile 64x32
// (4x2 WMMA accumulators). Double-buffered LDS; async global->LDS copies.
// grid = (N/128, M/128, batch)
// ---------------------------------------------------------------------------
__global__ void __launch_bounds__(256)
gemm_bf16_kernel(const u16* __restrict__ A,
                 const u16* __restrict__ BT,
                 float* __restrict__ D,
                 int M, int N, int K, float scale) {
    const int LDS_LD = 40;   // 32 + 8 pad (bf16 elems), rows stay 16B-aligned
    __shared__ __align__(16) u16 As[2][128 * LDS_LD];
    __shared__ __align__(16) u16 Bs[2][128 * LDS_LD];

    A  += (size_t)blockIdx.z * (size_t)M * (size_t)K;
    BT += (size_t)blockIdx.z * (size_t)N * (size_t)K;
    D  += (size_t)blockIdx.z * (size_t)M * (size_t)N;

    const int rowBase = blockIdx.y * 128;   // M
    const int colBase = blockIdx.x * 128;   // N

    const int lane  = threadIdx.x & 31;
    const int wave  = threadIdx.x >> 5;
    const int waveM = (wave & 1) * 64;      // 2 waves over M
    const int waveN = (wave >> 1) * 32;     // 4 waves over N
    const int mLane = lane & 15;
    const int kbA   = (lane >> 4) * 8;      // A frag: K halves {kbA..+7, kbA+16..+23}
    const int kbB   = (lane >> 4) * 8;      // B frag: K halves {kbB..+7, kbB+16..+23}

    // per-thread tile-copy coordinates (2 x 16B chunks per operand per tile)
    const int r0 = threadIdx.x >> 2;                  // chunk i=0: row
    const int c0 = (threadIdx.x & 3) << 3;            //           col
    const int r1 = (threadIdx.x + 256) >> 2;          // chunk i=1
    const int c1 = ((threadIdx.x + 256) & 3) << 3;

    v8f acc[4][2] = {};

    const int nT = K >> 5;                  // K / 32 tiles

    // ---- prologue: issue tile 0 into buffer 0 ----
    {
        const u16* a = A + (size_t)(rowBase + r0) * K + c0;
        const u16* b = BT + (size_t)(colBase + r0) * K + c0;
        copy16_g2l(a, &As[0][r0 * LDS_LD + c0]);
        copy16_g2l(b, &Bs[0][r0 * LDS_LD + c0]);
        const u16* a2 = A + (size_t)(rowBase + r1) * K + c1;
        const u16* b2 = BT + (size_t)(colBase + r1) * K + c1;
        copy16_g2l(a2, &As[0][r1 * LDS_LD + c1]);
        copy16_g2l(b2, &Bs[0][r1 * LDS_LD + c1]);
    }

    for (int t = 0; t < nT; ++t) {
        wait_async_copies();
        __syncthreads();                    // buf[t&1] ready; buf[(t+1)&1] free

        if (t + 1 < nT) {
            int k0 = (t + 1) << 5;
            int nb = (t + 1) & 1;
            const u16* a = A + (size_t)(rowBase + r0) * K + k0 + c0;
            const u16* b = BT + (size_t)(colBase + r0) * K + k0 + c0;
            copy16_g2l(a, &As[nb][r0 * LDS_LD + c0]);
            copy16_g2l(b, &Bs[nb][r0 * LDS_LD + c0]);
            const u16* a2 = A + (size_t)(rowBase + r1) * K + k0 + c1;
            const u16* b2 = BT + (size_t)(colBase + r1) * K + k0 + c1;
            copy16_g2l(a2, &As[nb][r1 * LDS_LD + c1]);
            copy16_g2l(b2, &Bs[nb][r1 * LDS_LD + c1]);
        }

        // ---- fragments + 8 WMMA on buf[t&1] ----
        const int cb = t & 1;
        FragBF16 af[4], bf[2];
#pragma unroll
        for (int mi = 0; mi < 4; ++mi) {
            const u16* p = &As[cb][(waveM + mi * 16 + mLane) * LDS_LD + kbA];
            af[mi].u4[0] = *(const U4*)p;
            af[mi].u4[1] = *(const U4*)(p + 16);
        }
#pragma unroll
        for (int ni = 0; ni < 2; ++ni) {
            const u16* p = &Bs[cb][(waveN + ni * 16 + mLane) * LDS_LD + kbB];
            bf[ni].u4[0] = *(const U4*)p;
            bf[ni].u4[1] = *(const U4*)(p + 16);
        }
#pragma unroll
        for (int mi = 0; mi < 4; ++mi)
#pragma unroll
            for (int ni = 0; ni < 2; ++ni)
                acc[mi][ni] = __builtin_amdgcn_wmma_f32_16x16x32_bf16(
                    false, af[mi].v, false, bf[ni].v,
                    (short)0, acc[mi][ni], false, false);
    }

    // ---- store: C/D layout -> VGPR r holds row (r | r+8), col = lane&15 ----
#pragma unroll
    for (int mi = 0; mi < 4; ++mi) {
#pragma unroll
        for (int ni = 0; ni < 2; ++ni) {
            int col  = colBase + waveN + ni * 16 + (lane & 15);
            int rtop = rowBase + waveM + mi * 16 + ((lane < 16) ? 0 : 8);
#pragma unroll
            for (int r = 0; r < 8; ++r)
                D[(size_t)(rtop + r) * N + col] = acc[mi][ni][r] * scale;
        }
    }
}

// ---------------------------------------------------------------------------
// Kernel 4: row softmax over 1024 f32 values -> bf16 attn. Block per row.
// ---------------------------------------------------------------------------
__global__ void softmax_rows_kernel(const float* __restrict__ S,
                                    u16* __restrict__ Aout) {
    const int Ncol = 1024;
    const float* row = S + (size_t)blockIdx.x * Ncol;
    u16* orow = Aout + (size_t)blockIdx.x * Ncol;

    __shared__ float red[256];

    float vals[4];
    float mx = -3.4e38f;
#pragma unroll
    for (int j = 0; j < 4; ++j) {
        vals[j] = row[threadIdx.x + j * 256];
        mx = fmaxf(mx, vals[j]);
    }
    red[threadIdx.x] = mx;
    __syncthreads();
    for (int s = 128; s > 0; s >>= 1) {
        if (threadIdx.x < s) red[threadIdx.x] = fmaxf(red[threadIdx.x], red[threadIdx.x + s]);
        __syncthreads();
    }
    mx = red[0];
    __syncthreads();

    float sum = 0.0f;
#pragma unroll
    for (int j = 0; j < 4; ++j) {
        vals[j] = __expf(vals[j] - mx);
        sum += vals[j];
    }
    red[threadIdx.x] = sum;
    __syncthreads();
    for (int s = 128; s > 0; s >>= 1) {
        if (threadIdx.x < s) red[threadIdx.x] += red[threadIdx.x + s];
        __syncthreads();
    }
    float inv = 1.0f / red[0];
#pragma unroll
    for (int j = 0; j < 4; ++j)
        orow[threadIdx.x + j * 256] = f2bf(vals[j] * inv);
}

// ---------------------------------------------------------------------------
// Host launcher
// ---------------------------------------------------------------------------
extern "C" void kernel_launch(void* const* d_in, const int* in_sizes, int n_in,
                              void* d_out, int out_size, void* d_ws, size_t ws_size,
                              hipStream_t stream) {
    const int C_ = 1024, L_ = 4096;
    const int B_ = in_sizes[0] / (C_ * L_);          // 8

    const float* x  = (const float*)d_in[0];
    const float* qw = (const float*)d_in[1];
    const float* kw = (const float*)d_in[2];
    const float* vw = (const float*)d_in[3];
    float* out = (float*)d_out;

    const size_t qkvElems = (size_t)B_ * C_ * L_;    // 33.5M elems (64MB bf16 each)
    u16*   q    = (u16*)d_ws;
    u16*   k    = q + qkvElems;
    u16*   v    = k + qkvElems;
    float* S    = (float*)(v + qkvElems);            // [B,C,C] f32 (32MB)
    u16*   vT   = q;                                 // alias q (dead after GEMM1)
    u16*   attn = k;                                 // alias k (dead after GEMM1)

    // 1) fused depthwise conv -> q,k,v bf16
    dwconv_qkv_kernel<<<dim3(B_ * C_), dim3(256), 0, stream>>>(x, qw, kw, vw, q, k, v);

    // 2) score = (q @ k^T) / sqrt(C): M=N=1024, K=4096 (both K-contiguous)
    dim3 g1(C_ / 128, C_ / 128, B_);
    gemm_bf16_kernel<<<g1, dim3(256), 0, stream>>>(q, k, S, C_, C_, L_, 0.03125f);

    // 3) v [C x L] -> vT [L x C] (per batch), into dead q region
    dim3 gt(L_ / 64, C_ / 64, B_);
    transpose_bf16_kernel<<<gt, dim3(256), 0, stream>>>(v, vT, C_, L_);

    // 4) softmax rows -> attn bf16 (into dead k region)
    softmax_rows_kernel<<<dim3(B_ * C_), dim3(256), 0, stream>>>(S, attn);

    // 5) out = attn @ v = attn @ vT^T: M=1024, N=4096, K=1024
    dim3 g2(L_ / 128, C_ / 128, B_);
    gemm_bf16_kernel<<<g2, dim3(256), 0, stream>>>(attn, vT, out, C_, L_, C_, 1.0f);
}
